// AttentionMem_79886391706062
// MI455X (gfx1250) — matrix operand order repaired and checked by reading, not compiled, not run on hardware
//
#include <hip/hip_runtime.h>

#define DIMC   512
#define NQ     4096
#define NBAT   2
#define NMEM   1024
#define NKV    5120          // NQ + NMEM
#define HEADS  8
#define DHEAD  64
#define NPEG   5119          // NKV - 1 (PEG sequence length)

typedef __attribute__((ext_vector_type(16))) __bf16 v16bf;
typedef __attribute__((ext_vector_type(8)))  __bf16 v8bf;
typedef __attribute__((ext_vector_type(8)))  float  v8f;
typedef __attribute__((ext_vector_type(4)))  unsigned int v4u;
typedef __attribute__((ext_vector_type(8)))  int v8i;
typedef __attribute__((ext_vector_type(4)))  int v4i;

#if __has_builtin(__builtin_amdgcn_tensor_load_to_lds) && \
    __has_builtin(__builtin_amdgcn_s_wait_tensorcnt)
#define HAVE_TDM 1
#else
#define HAVE_TDM 0
#endif

__device__ __forceinline__ __bf16 f2bf(float f) {
  unsigned u = __builtin_bit_cast(unsigned, f);
  u = (u + 0x7FFFu + ((u >> 16) & 1u)) >> 16;
  unsigned short s = (unsigned short)u;
  return __builtin_bit_cast(__bf16, s);
}

// ---------------- LayerNorm -> bf16 ----------------
__global__ __launch_bounds__(256) void ln_kernel(const float* __restrict__ x,
                                                 const float* __restrict__ g,
                                                 const float* __restrict__ bta,
                                                 __bf16* __restrict__ xn) {
  int row = blockIdx.x;            // 0..8191
  int tid = threadIdx.x;           // 256
  const float* xr = x + (size_t)row * DIMC;
  float v0 = xr[tid], v1 = xr[tid + 256];
  float s = v0 + v1, ss = v0 * v0 + v1 * v1;
  for (int o = 16; o > 0; o >>= 1) { s += __shfl_xor(s, o, 32); ss += __shfl_xor(ss, o, 32); }
  __shared__ float red[16];
  int wv = tid >> 5, lane = tid & 31;
  if (lane == 0) { red[wv] = s; red[8 + wv] = ss; }
  __syncthreads();
  float ts = 0.f, tss = 0.f;
  for (int w = 0; w < 8; w++) { ts += red[w]; tss += red[8 + w]; }
  float mu = ts * (1.f / DIMC);
  float var = tss * (1.f / DIMC) - mu * mu;
  float inv = rsqrtf(var + 1e-5f);
  xn[(size_t)row * DIMC + tid]       = f2bf((v0 - mu) * inv * g[tid]       + bta[tid]);
  xn[(size_t)row * DIMC + tid + 256] = f2bf((v1 - mu) * inv * g[tid + 256] + bta[tid + 256]);
}

// ------------- transpose fp32 weight -> bf16 (col-major) -------------
__global__ __launch_bounds__(256) void wtrans_kernel(const float* __restrict__ w,
                                                     __bf16* __restrict__ wt,
                                                     int rows, int cols) {
  int idx = blockIdx.x * 256 + threadIdx.x;
  if (idx < rows * cols) {
    int k = idx / cols, n = idx % cols;
    wt[(size_t)n * rows + k] = f2bf(w[idx]);
  }
}

// ------------- copy mem into K/V f32 tails -------------
__global__ __launch_bounds__(256) void memtail_kernel(const float* __restrict__ mem,
                                                      float* __restrict__ kf,
                                                      float* __restrict__ vf) {
  size_t idx = (size_t)blockIdx.x * 256 + threadIdx.x;   // 2*1024*512
  size_t tot = (size_t)NBAT * NMEM * DIMC;
  if (idx < tot) {
    size_t b = idx / ((size_t)NMEM * DIMC);
    size_t r = idx % ((size_t)NMEM * DIMC);
    size_t dst = ((size_t)b * NKV + NQ) * DIMC + r;
    float v = mem[idx];
    kf[dst] = v; vf[dst] = v;
  }
}

// ------ WMMA helper: one wave, 32x64 tile (2 M-subtiles reuse each B frag), K=512 ------
__device__ __forceinline__ void wmma_tile2_k512(const __bf16* __restrict__ A,
                                                const __bf16* __restrict__ Bt,
                                                int m0, int n0, int lane, v8f acc[2][4]) {
  int lh = lane & 15, ls = lane >> 4;
  for (int k0 = 0; k0 < DIMC; k0 += 32) {
    v16bf af[2];
#pragma unroll
    for (int t = 0; t < 2; t++) {
      const __bf16* ap = A + (size_t)(m0 + t * 16 + lh) * DIMC + k0 + ls * 8;
      ((v8bf*)&af[t])[0] = *(const v8bf*)ap;
      ((v8bf*)&af[t])[1] = *(const v8bf*)(ap + 16);
    }
#pragma unroll
    for (int j = 0; j < 4; j++) {
      const __bf16* bp = Bt + (size_t)(n0 + j * 16 + lh) * DIMC + k0 + ls * 8;
      v16bf bfr;
      ((v8bf*)&bfr)[0] = *(const v8bf*)bp;
      ((v8bf*)&bfr)[1] = *(const v8bf*)(bp + 16);
      acc[0][j] = __builtin_amdgcn_wmma_f32_16x16x32_bf16(false, af[0], false, bfr,
                                                          (short)0, acc[0][j], false, false);
      acc[1][j] = __builtin_amdgcn_wmma_f32_16x16x32_bf16(false, af[1], false, bfr,
                                                          (short)0, acc[1][j], false, false);
    }
  }
}

// ------------- QKV GEMM: xn(8192x512) @ wqkvT -> q(bf16), k/v(f32 at concat rows) ----
__global__ __launch_bounds__(256) void gemm_qkv_kernel(const __bf16* __restrict__ A,
                                                       const __bf16* __restrict__ Bt,
                                                       __bf16* __restrict__ qb,
                                                       float* __restrict__ kf,
                                                       float* __restrict__ vf) {
  int wave = threadIdx.x >> 5, lane = threadIdx.x & 31;
  int m0 = blockIdx.y * 64 + (wave >> 2) * 32;
  int n0 = blockIdx.x * 256 + (wave & 3) * 64;
  int lh = lane & 15, ls = lane >> 4;
  v8f acc[2][4] = {};
  wmma_tile2_k512(A, Bt, m0, n0, lane, acc);
#pragma unroll
  for (int t = 0; t < 2; t++) {
#pragma unroll
    for (int j = 0; j < 4; j++) {
      int n = n0 + j * 16 + lh;
#pragma unroll
      for (int r = 0; r < 8; r++) {
        int mrow = m0 + t * 16 + r + 8 * ls;
        int b = mrow >> 12, nn = mrow & 4095;
        float val = acc[t][j][r];
        if (n < 512)        qb[(size_t)mrow * DIMC + n] = f2bf(val);
        else if (n < 1024)  kf[((size_t)(b * NKV + nn)) * DIMC + (n - 512)]  = val;
        else                vf[((size_t)(b * NKV + nn)) * DIMC + (n - 1024)] = val;
      }
    }
  }
}

// ------------- PEG1D depthwise conv on K/V (f32 in, bf16 out) -------------
__global__ __launch_bounds__(256) void peg_kernel(const float* __restrict__ kf,
                                                  const float* __restrict__ vf,
                                                  const float* __restrict__ pkw,
                                                  const float* __restrict__ pkb,
                                                  const float* __restrict__ pvw,
                                                  const float* __restrict__ pvb,
                                                  __bf16* __restrict__ kb,
                                                  __bf16* __restrict__ vb) {
  size_t idx = (size_t)blockIdx.x * 256 + threadIdx.x;    // NBAT*NKV*DIMC
  size_t tot = (size_t)NBAT * NKV * DIMC;
  if (idx >= tot) return;
  int b = (int)(idx / ((size_t)NKV * DIMC));
  int rem = (int)(idx % ((size_t)NKV * DIMC));
  int pos = rem / DIMC, c = rem % DIMC;
  size_t self = ((size_t)(b * NKV + pos)) * DIMC + c;
  if (pos == 0) {                       // cls token passes through
    kb[self] = f2bf(kf[self]);
    vb[self] = f2bf(vf[self]);
    return;
  }
  int i = pos - 1;                      // 0..NPEG-1
  float wk0 = pkw[c * 3], wk1 = pkw[c * 3 + 1], wk2 = pkw[c * 3 + 2];
  float wv0 = pvw[c * 3], wv1 = pvw[c * 3 + 1], wv2 = pvw[c * 3 + 2];
  float sk = 2.f * pkb[c], sv = 2.f * pvb[c];   // bias from p1 and p2
  // p1: conv within chunk of 32 at position i
  {
    int chunk = i >> 5, t = i & 31;
#pragma unroll
    for (int dt = 0; dt < 3; dt++) {
      int tt = t + dt - 1;
      if (tt >= 0 && tt < 32) {
        int l = (chunk << 5) + tt;
        if (l < NPEG) {
          size_t off = ((size_t)(b * NKV + 1 + l)) * DIMC + c;
          float wkx = (dt == 0) ? wk0 : (dt == 1 ? wk1 : wk2);
          float wvx = (dt == 0) ? wv0 : (dt == 1 ? wv1 : wv2);
          sk += wkx * kf[off]; sv += wvx * vf[off];
        }
      }
    }
  }
  // p2: roll(+16), conv, roll(-16)  ->  conv at jj=(i+16)%NPEG on shifted seq
  {
    int jj = (i + 16) % NPEG;
    int chunk = jj >> 5, t = jj & 31;
#pragma unroll
    for (int dt = 0; dt < 3; dt++) {
      int tt = t + dt - 1;
      if (tt >= 0 && tt < 32) {
        int l = (chunk << 5) + tt;
        if (l < NPEG) {
          int src = l - 16; if (src < 0) src += NPEG;
          size_t off = ((size_t)(b * NKV + 1 + src)) * DIMC + c;
          float wkx = (dt == 0) ? wk0 : (dt == 1 ? wk1 : wk2);
          float wvx = (dt == 0) ? wv0 : (dt == 1 ? wv1 : wv2);
          sk += wkx * kf[off]; sv += wvx * vf[off];
        }
      }
    }
  }
  kb[self] = f2bf(kf[self] + sk);
  vb[self] = f2bf(vf[self] + sv);
}

// ------------- Flash attention: TDM-staged tiles + WMMA QK^T / PV -------------
__global__ __launch_bounds__(256) void attn_kernel(const __bf16* __restrict__ Q,
                                                   const __bf16* __restrict__ K,
                                                   const __bf16* __restrict__ V,
                                                   __bf16* __restrict__ O) {
  int wv = threadIdx.x >> 5, lane = threadIdx.x & 31;
  int h = blockIdx.y, b = blockIdx.z;
  int q0 = blockIdx.x * 128 + wv * 16;
  int lh = lane & 15, ls = lane >> 4;

  __shared__ alignas(32) __bf16 kt[32 * 64];        // K tile (m, d)
  __shared__ alignas(32) __bf16 vraw[32 * 64];      // V tile raw (m, d)
  __shared__ alignas(32) __bf16 vt[64 * 32];        // V tile transposed (d, m)
  __shared__ alignas(16) __bf16 ps[8 * 16 * 32];    // per-wave P scratch

  // Q A-fragments (16 rows x 64 d, split K=32 twice)
  v16bf qa[2];
  const __bf16* qbase = Q + ((size_t)(b * NQ + q0 + lh)) * DIMC + h * DHEAD;
#pragma unroll
  for (int kc = 0; kc < 2; kc++) {
    const __bf16* p = qbase + kc * 32 + ls * 8;
    ((v8bf*)&qa[kc])[0] = *(const v8bf*)p;
    ((v8bf*)&qa[kc])[1] = *(const v8bf*)(p + 16);
  }

  v8f o[4] = {};
  float mrow[8], lrow[8];
#pragma unroll
  for (int r = 0; r < 8; r++) { mrow[r] = -1e30f; lrow[r] = 0.f; }

  const __bf16* Kb = K + ((size_t)b * NKV) * DIMC + h * DHEAD;
  const __bf16* Vb = V + ((size_t)b * NKV) * DIMC + h * DHEAD;

#if HAVE_TDM
  // D# group1 (invariant): data_size=2B, tensor 64x32 tile 64x32, dim0 stride 512
  v8i g1;
  g1[0] = 0x00010000;            // wg_mask=0, data_size=1(2B), no barrier/iter/pad
  g1[1] = (int)(64u << 16);      // tensor_dim0 = 64 (bits 79:48, low half)
  g1[2] = (int)(32u << 16);      // tensor_dim0 hi=0, tensor_dim1 = 32 (low 16)
  g1[3] = (int)(64u << 16);      // tensor_dim1 hi=0, tile_dim0 = 64
  g1[4] = 32;                    // tile_dim1 = 32, tile_dim2 = 0
  g1[5] = 512;                   // tensor_dim0_stride = 512 (low 32)
  g1[6] = 0;                     // stride hi, tensor_dim1_stride lo
  g1[7] = 0;
  v4i gz = {0, 0, 0, 0};
  v8i gz8 = {0, 0, 0, 0, 0, 0, 0, 0};
  unsigned lds_k = (unsigned)(size_t)&kt[0];
  unsigned lds_v = (unsigned)(size_t)&vraw[0];
#endif

  for (int m0 = 0; m0 < NKV; m0 += 32) {
#if HAVE_TDM
    if (wv == 0) {
      unsigned long long gak = (unsigned long long)(size_t)(Kb + (size_t)m0 * DIMC);
      unsigned long long gav = (unsigned long long)(size_t)(Vb + (size_t)m0 * DIMC);
      v4u g0;
      g0[0] = 1u;                                    // count=1, user mode
      g0[1] = lds_k;
      g0[2] = (unsigned)gak;
      g0[3] = 0x80000000u | (unsigned)(gak >> 32);   // type=2 ("image") | addr hi
      __builtin_amdgcn_tensor_load_to_lds(g0, g1, gz, gz, gz8, 0);
      g0[1] = lds_v;
      g0[2] = (unsigned)gav;
      g0[3] = 0x80000000u | (unsigned)(gav >> 32);
      __builtin_amdgcn_tensor_load_to_lds(g0, g1, gz, gz, gz8, 0);
      __builtin_amdgcn_s_wait_tensorcnt(0);
    }
    __syncthreads();
    // cooperative transpose: vraw(m,d) -> vt(d,m)
    {
      int mi = threadIdx.x >> 3;
      int d0 = (threadIdx.x & 7) << 3;
      v8bf vvv = *(const v8bf*)&vraw[mi * 64 + d0];
#pragma unroll
      for (int e = 0; e < 8; e++) vt[(d0 + e) * 32 + mi] = vvv[e];
      if (m0 + 32 < NKV) {
        __builtin_prefetch(Kb + (size_t)(m0 + 32 + mi) * DIMC + d0, 0, 1);
        __builtin_prefetch(Vb + (size_t)(m0 + 32 + mi) * DIMC + d0, 0, 1);
      }
    }
    __syncthreads();
#else
    // fallback: manual global->LDS staging
    {
      int mi = threadIdx.x >> 3;
      int d0 = (threadIdx.x & 7) << 3;
      const __bf16* kp = Kb + (size_t)(m0 + mi) * DIMC + d0;
      const __bf16* vp = Vb + (size_t)(m0 + mi) * DIMC + d0;
      v8bf kvv = *(const v8bf*)kp;
      *(v8bf*)&kt[mi * 64 + d0] = kvv;
      v8bf vvv = *(const v8bf*)vp;
#pragma unroll
      for (int e = 0; e < 8; e++) vt[(d0 + e) * 32 + mi] = vvv[e];
      if (m0 + 32 < NKV) {
        __builtin_prefetch(kp + 32 * DIMC, 0, 1);
        __builtin_prefetch(vp + 32 * DIMC, 0, 1);
      }
    }
    __syncthreads();
#endif

    // S = Q K^T  (two 16x16 tiles over 32 m-columns, K-dim 64)
    v8f s[2] = {};
#pragma unroll
    for (int j = 0; j < 2; j++) {
      const __bf16* p0 = &kt[(j * 16 + lh) * 64 + ls * 8];
      v16bf kb0, kb1;
      ((v8bf*)&kb0)[0] = *(const v8bf*)p0;
      ((v8bf*)&kb0)[1] = *(const v8bf*)(p0 + 16);
      ((v8bf*)&kb1)[0] = *(const v8bf*)(p0 + 32);
      ((v8bf*)&kb1)[1] = *(const v8bf*)(p0 + 48);
      s[j] = __builtin_amdgcn_wmma_f32_16x16x32_bf16(false, qa[0], false, kb0,
                                                     (short)0, s[j], false, false);
      s[j] = __builtin_amdgcn_wmma_f32_16x16x32_bf16(false, qa[1], false, kb1,
                                                     (short)0, s[j], false, false);
    }

    // online softmax, row stats replicated across each 16-lane half
    float alpha[8];
#pragma unroll
    for (int r = 0; r < 8; r++) {
      float a0 = s[0][r] * 0.125f, a1 = s[1][r] * 0.125f;
      float tmax = fmaxf(a0, a1);
      for (int off = 8; off > 0; off >>= 1) tmax = fmaxf(tmax, __shfl_xor(tmax, off, 16));
      float mnew = fmaxf(mrow[r], tmax);
      float sc = __expf(mrow[r] - mnew);
      float p0 = __expf(a0 - mnew), p1 = __expf(a1 - mnew);
      float rs = p0 + p1;
      for (int off = 8; off > 0; off >>= 1) rs += __shfl_xor(rs, off, 16);
      lrow[r] = lrow[r] * sc + rs;
      mrow[r] = mnew;
      alpha[r] = sc;
      int row = r + 8 * ls;             // C-layout row -> (row, m) scratch
      ps[wv * 512 + row * 32 + lh]      = f2bf(p0);
      ps[wv * 512 + row * 32 + 16 + lh] = f2bf(p1);
    }
#pragma unroll
    for (int dj = 0; dj < 4; dj++)
#pragma unroll
      for (int r = 0; r < 8; r++) o[dj][r] *= alpha[r];

    asm volatile("s_wait_dscnt 0" ::: "memory");    // P stores visible to own wave

    // P A-fragment from scratch (row = lane%16, m split by lane half)
    v16bf pa;
    {
      const __bf16* pp = &ps[wv * 512 + lh * 32 + ls * 8];
      ((v8bf*)&pa)[0] = *(const v8bf*)pp;
      ((v8bf*)&pa)[1] = *(const v8bf*)(pp + 16);
    }
    // O += P V  (4 d-blocks of 16)
#pragma unroll
    for (int dj = 0; dj < 4; dj++) {
      v16bf vbf;
      ((v8bf*)&vbf)[0] = *(const v8bf*)&vt[(dj * 16 + lh) * 32 + ls * 8];
      ((v8bf*)&vbf)[1] = *(const v8bf*)&vt[(dj * 16 + lh) * 32 + ls * 8 + 16];
      o[dj] = __builtin_amdgcn_wmma_f32_16x16x32_bf16(false, pa, false, vbf,
                                                      (short)0, o[dj], false, false);
    }
    __syncthreads();
  }

  // normalize + store bf16 at (b, n, h*64+d)
#pragma unroll
  for (int r = 0; r < 8; r++) {
    float inv = 1.f / lrow[r];
    int n = q0 + r + 8 * ls;
#pragma unroll
    for (int dj = 0; dj < 4; dj++)
      O[((size_t)(b * NQ + n)) * DIMC + h * DHEAD + dj * 16 + lh] = f2bf(o[dj][r] * inv);
  }
}

// ------------- Output projection GEMM + bias -> f32 -------------
__global__ __launch_bounds__(256) void gemm_out_kernel(const __bf16* __restrict__ A,
                                                       const __bf16* __restrict__ Bt,
                                                       const float* __restrict__ bias,
                                                       float* __restrict__ out) {
  int wave = threadIdx.x >> 5, lane = threadIdx.x & 31;
  int m0 = blockIdx.y * 64 + (wave >> 2) * 32;
  int n0 = blockIdx.x * 256 + (wave & 3) * 64;
  int lh = lane & 15, ls = lane >> 4;
  v8f acc[2][4] = {};
  wmma_tile2_k512(A, Bt, m0, n0, lane, acc);
#pragma unroll
  for (int t = 0; t < 2; t++) {
#pragma unroll
    for (int j = 0; j < 4; j++) {
      int n = n0 + j * 16 + lh;
      float bv = bias[n];
#pragma unroll
      for (int r = 0; r < 8; r++) {
        int mrow = m0 + t * 16 + r + 8 * ls;
        out[(size_t)mrow * DIMC + n] = acc[t][j][r] + bv;
      }
    }
  }
}

extern "C" void kernel_launch(void* const* d_in, const int* in_sizes, int n_in,
                              void* d_out, int out_size, void* d_ws, size_t ws_size,
                              hipStream_t stream) {
  const float* x     = (const float*)d_in[0];
  const float* mem   = (const float*)d_in[1];
  const float* ln_g  = (const float*)d_in[2];
  const float* ln_b  = (const float*)d_in[3];
  const float* w_qkv = (const float*)d_in[4];
  const float* w_out = (const float*)d_in[5];
  const float* b_out = (const float*)d_in[6];
  const float* pk_w  = (const float*)d_in[7];
  const float* pk_b  = (const float*)d_in[8];
  const float* pv_w  = (const float*)d_in[9];
  const float* pv_b  = (const float*)d_in[10];
  float* out = (float*)d_out;

  char* ws = (char*)d_ws;
  __bf16* xn    = (__bf16*)ws; ws += (size_t)8192 * 512 * 2;
  __bf16* wqkvT = (__bf16*)ws; ws += (size_t)1536 * 512 * 2;
  __bf16* woutT = (__bf16*)ws; ws += (size_t)512 * 512 * 2;
  __bf16* qb    = (__bf16*)ws; ws += (size_t)8192 * 512 * 2;
  float*  kf    = (float*)ws;  ws += (size_t)NBAT * NKV * 512 * 4;
  float*  vf    = (float*)ws;  ws += (size_t)NBAT * NKV * 512 * 4;
  __bf16* kb    = (__bf16*)ws; ws += (size_t)NBAT * NKV * 512 * 2;
  __bf16* vb    = (__bf16*)ws; ws += (size_t)NBAT * NKV * 512 * 2;
  __bf16* ao    = (__bf16*)ws; ws += (size_t)8192 * 512 * 2;

  wtrans_kernel<<<(512 * 1536 + 255) / 256, 256, 0, stream>>>(w_qkv, wqkvT, 512, 1536);
  wtrans_kernel<<<(512 * 512 + 255) / 256, 256, 0, stream>>>(w_out, woutT, 512, 512);
  ln_kernel<<<8192, 256, 0, stream>>>(x, ln_g, ln_b, xn);
  memtail_kernel<<<(NBAT * NMEM * 512 + 255) / 256, 256, 0, stream>>>(mem, kf, vf);
  gemm_qkv_kernel<<<dim3(6, 128), 256, 0, stream>>>(xn, wqkvT, qb, kf, vf);
  peg_kernel<<<(NBAT * NKV * 512 + 255) / 256, 256, 0, stream>>>(kf, vf, pk_w, pk_b,
                                                                 pv_w, pv_b, kb, vb);
  attn_kernel<<<dim3(NQ / 128, HEADS, NBAT), 256, 0, stream>>>(qb, kb, vb, ao);
  gemm_out_kernel<<<dim3(2, 128), 256, 0, stream>>>(ao, woutT, b_out, out);
}
